// BlockAxialUp_9560597201242
// MI455X (gfx1250) — matrix operand order repaired and checked by reading, not compiled, not run on hardware
//
#include <hip/hip_runtime.h>

typedef __attribute__((ext_vector_type(16))) _Float16 v16h;
typedef __attribute__((ext_vector_type(8)))  float    v8f;

union V16H { v16h v; uint4 q[2]; _Float16 h[16]; };

__device__ __forceinline__ v8f vzero8() {
  v8f z;
#pragma unroll
  for (int i = 0; i < 8; ++i) z[i] = 0.f;
  return z;
}

__device__ __forceinline__ v8f wmma_f16(v16h a, v16h b, v8f c) {
  // D = A(16x32 f16) * B(32x16 f16) + C(16x16 f32)
  return __builtin_amdgcn_wmma_f32_16x16x32_f16(false, a, false, b, (short)0, c, false, false);
}

// A-fragment (16x32). rowk0 = &T[m][k0], row-major, K contiguous.
// elements 0..7  <- K = k0 + 8*hi + (0..7)
// elements 8..15 <- K = k0 + 16 + 8*hi + (0..7)
__device__ __forceinline__ v16h load_frag_a(const _Float16* rowk0, int hi) {
  V16H r;
  r.q[0] = *(const uint4*)(rowk0 + 8 * hi);
  r.q[1] = *(const uint4*)(rowk0 + 16 + 8 * hi);
  return r.v;
}

// B-fragment (32x16) where the source is stored "transposed": rowk0 = &T[n][k0]
// element j <- B[k][n] = T[n][k], k = k0 + 8*hi + j (j<8), k0 + 16 + 8*hi + (j-8) (j>=8)
__device__ __forceinline__ v16h load_frag_bt(const _Float16* rowk0, int hi) {
  V16H r;
  r.q[0] = *(const uint4*)(rowk0 + 8 * hi);
  r.q[1] = *(const uint4*)(rowk0 + 16 + 8 * hi);
  return r.v;
}

// B-fragment from pre-packed weight buffer: [(tile)*32 + lane]*16 halfs, contiguous.
__device__ __forceinline__ v16h load_frag_w(const _Float16* wbuf, int tile, int lane) {
  const uint4* p = (const uint4*)(wbuf + ((size_t)(tile * 32 + lane) << 4));
  V16H r;
  r.q[0] = p[0];
  r.q[1] = p[1];
  return r.v;
}

// gfx1250 async global->LDS copy (16B), tracked by ASYNCcnt.
// GVS mode: mem_addr = SGPR base + 32-bit VGPR byte offset; VDST = LDS byte addr.
__device__ __forceinline__ void async_g2l_b128(void* lds_ptr, const void* gbase,
                                               unsigned gbyte_off) {
  unsigned lds_addr = (unsigned)(uintptr_t)lds_ptr;  // low 32 bits = LDS offset
  asm volatile("global_load_async_to_lds_b128 %0, %1, %2"
               :: "v"(lds_addr), "v"(gbyte_off), "s"(gbase)
               : "memory");
}
__device__ __forceinline__ void wait_async0() {
  asm volatile("s_wait_asynccnt 0x0" ::: "memory");
}

// ---------------------------------------------------------------------------
// Weight pre-pack: f32 row-major (K x N) -> f16 fragment layout.
// dst[((nt*KT + kt)*32 + lane)*16 + j] = src[(kt*32 + (lane>>4)*8 + (j&7) + (j>>3)*16)*N + nt*16 + (lane&15)]
// ---------------------------------------------------------------------------
__global__ void wxform(const float* __restrict__ src, _Float16* __restrict__ dst,
                       int K, int N) {
  int idx = blockIdx.x * 256 + threadIdx.x;
  if (idx >= K * N) return;
  int j    = idx & 15;
  int lane = (idx >> 4) & 31;
  int tile = idx >> 9;
  int KT   = K >> 5;
  int kt = tile % KT, nt = tile / KT;
  int row = kt * 32 + (lane >> 4) * 8 + (j & 7) + ((j >> 3) << 4);
  int col = nt * 16 + (lane & 15);
  dst[idx] = (_Float16)src[row * N + col];
}

// ---------------------------------------------------------------------------
// Upsample 2x bilinear align_corners: x (4,256,64,64) f32 NCHW -> xu NHWC f16 (4,128,128,256)
// ---------------------------------------------------------------------------
__global__ void __launch_bounds__(256)
upsample2x(const float* __restrict__ x, _Float16* __restrict__ xu) {
  extern __shared__ _Float16 tile[];  // 128*256 halfs = 64 KB
  const int h = blockIdx.x, b = blockIdx.y;
  const int t = threadIdx.x;
  const int w = t & 127, cx = t >> 7;
  float chf = h * (63.f / 127.f);
  int i0 = (int)floorf(chf); if (i0 > 62) i0 = 62;
  float wh = chf - (float)i0;
  float cwf = w * (63.f / 127.f);
  int j0 = (int)floorf(cwf); if (j0 > 62) j0 = 62;
  float ww = cwf - (float)j0;
  for (int ci = 0; ci < 128; ++ci) {
    int c = ci * 2 + cx;
    const float* p = x + (((size_t)b * 256 + c) * 64 + i0) * 64 + j0;
    float v00 = p[0], v01 = p[1], v10 = p[64], v11 = p[65];
    float v = (v00 * (1.f - ww) + v01 * ww) * (1.f - wh) +
              (v10 * (1.f - ww) + v11 * ww) * wh;
    tile[w * 256 + c] = (_Float16)v;
  }
  __syncthreads();
  uint4* dst = (uint4*)(xu + (((size_t)b * 128 + h) << 7) * 256);
  const uint4* src = (const uint4*)tile;
  for (int i = t; i < 4096; i += 256) dst[i] = src[i];
}

// res (4,128,128,128) f32 NCHW -> NHWC f16
__global__ void __launch_bounds__(256)
res_to_nhwc(const float* __restrict__ res, _Float16* __restrict__ resf) {
  __shared__ _Float16 tile[128 * 128];  // 32 KB
  const int h = blockIdx.x, b = blockIdx.y;
  const int t = threadIdx.x;
  const int w = t & 127, ex = t >> 7;
  for (int ei = 0; ei < 64; ++ei) {
    int e = ei * 2 + ex;
    tile[w * 128 + e] = (_Float16)res[(((size_t)b * 128 + e) * 128 + h) * 128 + w];
  }
  __syncthreads();
  uint4* dst = (uint4*)(resf + (((size_t)b * 128 + h) << 7) * 128);
  const uint4* src = (const uint4*)tile;
  for (int i = t; i < 2048; i += 256) dst[i] = src[i];
}

// ---------------------------------------------------------------------------
// Axial self-attention. One workgroup (8 wave32) per sequence (512 sequences).
// T=128, C=256, 2 heads of e=128. All tiles staged in LDS (~300 KB).
// axis==0: sequence over H (batch b,w); axis==1: sequence over W (batch b,h).
// ---------------------------------------------------------------------------
#define LDC        264   // 256 + 8 halfs pad
#define LDS_SMALL  136   // 128 + 8 halfs pad
#define OFF_XS     0
#define OFF_QS     33792
#define OFF_KS     67584
#define OFF_VT     101376
#define OFF_OH     136192
#define ATTN_LDS_HALFS 153600  // 307200 bytes

__global__ void __launch_bounds__(256)
attn_axial(const _Float16* __restrict__ xu,
           const _Float16* __restrict__ wqf,
           const _Float16* __restrict__ wkvf,
           const _Float16* __restrict__ wof,
           const float* __restrict__ wob,
           _Float16* __restrict__ outbuf,
           int axis) {
  extern __shared__ _Float16 smem[];
  _Float16* Xs  = smem + OFF_XS;   // 128 x LDC (input; reused as scores/probs)
  _Float16* Qs  = smem + OFF_QS;   // 128 x LDC
  _Float16* Ks  = smem + OFF_KS;   // 128 x LDC
  _Float16* Vt  = smem + OFF_VT;   // 256 x LDS_SMALL (V transposed: [c][key])
  _Float16* Oh  = smem + OFF_OH;   // 128 x LDS_SMALL (per-head attn output)
  _Float16* S16 = Xs;              // 128 x LDS_SMALL scores/probs (overlay)

  const int tid  = threadIdx.x;
  const int wv   = tid >> 5;
  const int lane = tid & 31;
  const int hi   = lane >> 4;
  const int ln   = lane & 15;
  const int m0   = wv << 4;
  const int s    = blockIdx.x;

  size_t base0; int rstride;
  if (axis == 0) {  // rows along H; fixed (b, w)
    base0 = ((size_t)(s >> 7) * 16384 + (size_t)(s & 127)) * 256;
    rstride = 32768;  // W*C
  } else {          // rows along W; fixed (b, h)
    base0 = (size_t)s * 32768;
    rstride = 256;    // C
  }

  // Warm GL2 with the weight fragments this block is about to stream.
  __builtin_prefetch(wqf + ((size_t)tid << 6), 0, 1);
  __builtin_prefetch(wkvf + ((size_t)tid << 7), 0, 1);

  // Stage X (128 x 256 f16) with async global->LDS copies (ASYNCcnt path).
  for (int id = tid; id < 4096; id += 256) {
    int tr = id >> 5, tc = (id & 31) << 3;
    async_g2l_b128(Xs + tr * LDC + tc, xu,
                   (unsigned)((base0 + (size_t)tr * rstride + tc) * 2));
  }
  wait_async0();
  __syncthreads();

  // Q, K, V GEMMs: M=128, K=256, N = 256(Q) + 256(K) + 256(V)
  for (int nt = 0; nt < 48; ++nt) {
    v8f acc = vzero8();
    const _Float16* wsrc = (nt < 16) ? wqf : wkvf;
    int wtile = (nt < 16) ? nt * 8 : (nt - 16) * 8;
#pragma unroll
    for (int kt = 0; kt < 8; ++kt) {
      v16h a = load_frag_a(&Xs[(m0 + ln) * LDC + kt * 32], hi);
      v16h b = load_frag_w(wsrc, wtile + kt, lane);
      acc = wmma_f16(a, b, acc);
    }
    if (nt < 16) {
#pragma unroll
      for (int i = 0; i < 8; ++i)
        Qs[(m0 + i + 8 * hi) * LDC + nt * 16 + ln] = (_Float16)acc[i];
    } else if (nt < 32) {
#pragma unroll
      for (int i = 0; i < 8; ++i)
        Ks[(m0 + i + 8 * hi) * LDC + (nt - 16) * 16 + ln] = (_Float16)acc[i];
    } else {  // V stored transposed: Vt[c][key]
#pragma unroll
      for (int i = 0; i < 8; ++i)
        Vt[((nt - 32) * 16 + ln) * LDS_SMALL + m0 + i + 8 * hi] = (_Float16)acc[i];
    }
  }
  __syncthreads();

  const float scale = 0.08838834764831845f;  // 128^-0.5
  v8f outacc[16];
#pragma unroll
  for (int nt = 0; nt < 16; ++nt) outacc[nt] = vzero8();

  for (int hd = 0; hd < 2; ++hd) {
    // scores = Q_h * K_h^T * scale  (128x128, K-dim = 128)
    for (int nt = 0; nt < 8; ++nt) {
      v8f acc = vzero8();
#pragma unroll
      for (int kt = 0; kt < 4; ++kt) {
        v16h a = load_frag_a(&Qs[(m0 + ln) * LDC + hd * 128 + kt * 32], hi);
        v16h b = load_frag_bt(&Ks[(nt * 16 + ln) * LDC + hd * 128 + kt * 32], hi);
        acc = wmma_f16(a, b, acc);
      }
#pragma unroll
      for (int i = 0; i < 8; ++i)
        S16[(m0 + i + 8 * hi) * LDS_SMALL + nt * 16 + ln] =
            (_Float16)(acc[i] * scale);
    }
    __syncthreads();

    // row softmax (one thread per row)
    if (tid < 128) {
      _Float16* row = &S16[tid * LDS_SMALL];
      float mx = -1e30f;
      for (int j = 0; j < 128; ++j) mx = fmaxf(mx, (float)row[j]);
      float sum = 0.f;
      for (int j = 0; j < 128; ++j) {
        float e = __expf((float)row[j] - mx);
        sum += e;
        row[j] = (_Float16)e;
      }
      float inv = 1.f / sum;
      for (int j = 0; j < 128; ++j) row[j] = (_Float16)((float)row[j] * inv);
    }
    __syncthreads();

    // O_h = P * V_h  (128x128, K-dim = 128 keys)
    for (int nt = 0; nt < 8; ++nt) {
      v8f acc = vzero8();
#pragma unroll
      for (int kt = 0; kt < 4; ++kt) {
        v16h a = load_frag_a(&S16[(m0 + ln) * LDS_SMALL + kt * 32], hi);
        v16h b = load_frag_bt(&Vt[(hd * 128 + nt * 16 + ln) * LDS_SMALL + kt * 32], hi);
        acc = wmma_f16(a, b, acc);
      }
#pragma unroll
      for (int i = 0; i < 8; ++i)
        Oh[(m0 + i + 8 * hi) * LDS_SMALL + nt * 16 + ln] = (_Float16)acc[i];
    }
    __syncthreads();

    // out += O_h * Wo[hd*128 : hd*128+128, :]
    for (int nt = 0; nt < 16; ++nt) {
#pragma unroll
      for (int kt = 0; kt < 4; ++kt) {
        v16h a = load_frag_a(&Oh[(m0 + ln) * LDS_SMALL + kt * 32], hi);
        v16h b = load_frag_w(wof, nt * 8 + hd * 4 + kt, lane);
        outacc[nt] = wmma_f16(a, b, outacc[nt]);
      }
    }
    __syncthreads();
  }

  // + bias, store f16 NHWC
  for (int nt = 0; nt < 16; ++nt) {
    float bias = wob[nt * 16 + ln];
#pragma unroll
    for (int i = 0; i < 8; ++i) {
      int t = m0 + i + 8 * hi;
      outbuf[base0 + (size_t)t * rstride + nt * 16 + ln] =
          (_Float16)(outacc[nt][i] + bias);
    }
  }
}

// ---------------------------------------------------------------------------
// Fused: x_attn = relu(ah+aw); cat GEMM (K=640) -> relu -> d_out (NCHW f32)
// + deterministic per-workgroup BN partial sums.
// ---------------------------------------------------------------------------
__global__ void __launch_bounds__(256)
conv_bn(const _Float16* __restrict__ ah, const _Float16* __restrict__ aw,
        const _Float16* __restrict__ xu, const _Float16* __restrict__ resf,
        const _Float16* __restrict__ wconv,
        float* __restrict__ out, float* __restrict__ partial) {
  __shared__ float red[2][8][2][8][16];  // 16 KB
  const int tid = threadIdx.x;
  const int wv = tid >> 5, lane = tid & 31, hi = lane >> 4, ln = lane & 15;
  const int m0 = blockIdx.x * 128 + (wv << 4);

  __builtin_prefetch(wconv + ((size_t)tid << 6), 0, 1);

  v8f acc[8];
#pragma unroll
  for (int nt = 0; nt < 8; ++nt) acc[nt] = vzero8();

  for (int kt = 0; kt < 20; ++kt) {
    v16h a;
    int row = m0 + ln;
    if (kt < 8) {  // relu(ah+aw) fragment built on the fly
      const _Float16* pa = ah + (size_t)row * 256 + kt * 32;
      const _Float16* pb = aw + (size_t)row * 256 + kt * 32;
      V16H xa, xb, r;
      xa.q[0] = *(const uint4*)(pa + 8 * hi);
      xa.q[1] = *(const uint4*)(pa + 16 + 8 * hi);
      xb.q[0] = *(const uint4*)(pb + 8 * hi);
      xb.q[1] = *(const uint4*)(pb + 16 + 8 * hi);
#pragma unroll
      for (int i = 0; i < 16; ++i) {
        float v = (float)xa.h[i] + (float)xb.h[i];
        r.h[i] = (_Float16)fmaxf(v, 0.f);
      }
      a = r.v;
    } else if (kt < 16) {
      a = load_frag_a(xu + (size_t)row * 256 + (kt - 8) * 32, hi);
    } else {
      a = load_frag_a(resf + (size_t)row * 128 + (kt - 16) * 32, hi);
    }
#pragma unroll
    for (int nt = 0; nt < 8; ++nt) {
      v16h b = load_frag_w(wconv, nt * 20 + kt, lane);
      acc[nt] = wmma_f16(a, b, acc[nt]);
    }
  }

#pragma unroll
  for (int nt = 0; nt < 8; ++nt) {
    int e = nt * 16 + ln;
    float ps = 0.f, pq = 0.f;
#pragma unroll
    for (int i = 0; i < 8; ++i) {
      int row = m0 + i + 8 * hi;
      int b = row >> 14, hw = row & 16383;
      float v = fmaxf(acc[nt][i], 0.f);
      out[((size_t)(b * 128 + e) << 14) + hw] = v;
      ps += v;
      pq += v * v;
    }
    red[0][wv][hi][nt][ln] = ps;
    red[1][wv][hi][nt][ln] = pq;
  }
  __syncthreads();
  if (tid < 128) {  // fixed-order (deterministic) within-workgroup reduce
    int nt = tid >> 4, l = tid & 15;
    float s = 0.f, q = 0.f;
#pragma unroll
    for (int w = 0; w < 8; ++w)
#pragma unroll
      for (int h2 = 0; h2 < 2; ++h2) {
        s += red[0][w][h2][nt][l];
        q += red[1][w][h2][nt][l];
      }
    partial[blockIdx.x * 128 + tid] = s;
    partial[512 * 128 + blockIdx.x * 128 + tid] = q;
  }
}

// Final BN: tree-reduce 512 partials per channel, normalize d_out in place.
__global__ void __launch_bounds__(256)
bn_final(float* __restrict__ out, const float* __restrict__ partial,
         const float* __restrict__ gamma, const float* __restrict__ beta) {
  __shared__ float s1[256], s2[256];
  const int e = blockIdx.x, tid = threadIdx.x;
  float a = 0.f, b = 0.f;
  for (int i = tid; i < 512; i += 256) {
    a += partial[i * 128 + e];
    b += partial[512 * 128 + i * 128 + e];
  }
  s1[tid] = a;
  s2[tid] = b;
  __syncthreads();
  for (int off = 128; off > 0; off >>= 1) {
    if (tid < off) { s1[tid] += s1[tid + off]; s2[tid] += s2[tid + off]; }
    __syncthreads();
  }
  const float N = 65536.f;
  float mu  = s1[0] / N;
  float var = s2[0] / N - mu * mu;
  float inv = rsqrtf(var + 1e-5f);
  float g  = gamma[e] * inv;
  float bt = beta[e] - mu * g;
  for (int idx = tid; idx < 65536; idx += 256) {
    int bb = idx >> 14, hw = idx & 16383;
    size_t p = ((size_t)(bb * 128 + e) << 14) + hw;
    out[p] = out[p] * g + bt;
  }
}

// ---------------------------------------------------------------------------
extern "C" void kernel_launch(void* const* d_in, const int* in_sizes, int n_in,
                              void* d_out, int out_size, void* d_ws, size_t ws_size,
                              hipStream_t stream) {
  (void)in_sizes; (void)n_in; (void)out_size; (void)ws_size;
  const float* x      = (const float*)d_in[0];
  const float* res    = (const float*)d_in[1];
  const float* wq_h   = (const float*)d_in[2];
  const float* wkv_h  = (const float*)d_in[3];
  const float* wo_h   = (const float*)d_in[4];
  const float* wob_h  = (const float*)d_in[5];
  const float* wq_w   = (const float*)d_in[6];
  const float* wkv_w  = (const float*)d_in[7];
  const float* wo_w   = (const float*)d_in[8];
  const float* wob_w  = (const float*)d_in[9];
  const float* conv_w = (const float*)d_in[10];
  const float* gamma  = (const float*)d_in[11];
  const float* beta   = (const float*)d_in[12];
  float* out = (float*)d_out;

  char* ws = (char*)d_ws;
  _Float16* XU   = (_Float16*)(ws + 0);          // 33,554,432 B
  _Float16* RESF = (_Float16*)(ws + 33554432);   // 16,777,216 B
  _Float16* AH   = (_Float16*)(ws + 50331648);   // 33,554,432 B
  _Float16* AW   = (_Float16*)(ws + 83886080);   // 33,554,432 B
  _Float16* WQH  = (_Float16*)(ws + 117440512);  // 131,072 B
  _Float16* WKVH = (_Float16*)(ws + 117571584);  // 262,144 B
  _Float16* WOH  = (_Float16*)(ws + 117833728);  // 131,072 B
  _Float16* WQW  = (_Float16*)(ws + 117964800);  // 131,072 B
  _Float16* WKVW = (_Float16*)(ws + 118095872);  // 262,144 B
  _Float16* WOW  = (_Float16*)(ws + 118358016);  // 131,072 B
  _Float16* WCV  = (_Float16*)(ws + 118489088);  // 163,840 B
  float*    PART = (float*)(ws + 118652928);     // 524,288 B

  (void)hipFuncSetAttribute((const void*)attn_axial,
                            hipFuncAttributeMaxDynamicSharedMemorySize,
                            ATTN_LDS_HALFS * 2);
  (void)hipFuncSetAttribute((const void*)upsample2x,
                            hipFuncAttributeMaxDynamicSharedMemorySize,
                            128 * 256 * 2);

  // Weight pre-pack into WMMA B-fragment layout (f16)
  wxform<<<256, 256, 0, stream>>>(wq_h, WQH, 256, 256);
  wxform<<<512, 256, 0, stream>>>(wkv_h, WKVH, 256, 512);
  wxform<<<256, 256, 0, stream>>>(wo_h, WOH, 256, 256);
  wxform<<<256, 256, 0, stream>>>(wq_w, WQW, 256, 256);
  wxform<<<512, 256, 0, stream>>>(wkv_w, WKVW, 256, 512);
  wxform<<<256, 256, 0, stream>>>(wo_w, WOW, 256, 256);
  wxform<<<320, 256, 0, stream>>>(conv_w, WCV, 640, 128);

  upsample2x<<<dim3(128, 4), 256, 128 * 256 * 2, stream>>>(x, XU);
  res_to_nhwc<<<dim3(128, 4), 256, 0, stream>>>(res, RESF);

  attn_axial<<<512, 256, ATTN_LDS_HALFS * 2, stream>>>(XU, WQH, WKVH, WOH, wob_h, AH, 0);
  attn_axial<<<512, 256, ATTN_LDS_HALFS * 2, stream>>>(XU, WQW, WKVW, WOW, wob_w, AW, 1);

  conv_bn<<<512, 256, 0, stream>>>(AH, AW, XU, RESF, WCV, out, PART);
  bn_final<<<128, 256, 0, stream>>>(out, PART, gamma, beta);
}
